// MultiRNNCell_78864189489880
// MI455X (gfx1250) — matrix operand order repaired and checked by reading, not compiled, not run on hardware
//
#include <hip/hip_runtime.h>
#include <hip/hip_bf16.h>

#define B_  256
#define T_  256
#define DIN 128
#define DH  512

typedef __attribute__((ext_vector_type(16))) __bf16 v16bf;
typedef __attribute__((ext_vector_type(8)))  __bf16 v8bf;
typedef __attribute__((ext_vector_type(8)))  float  v8f;

__device__ __forceinline__ v8f wmma_bf(v16bf a, v16bf b, v8f c) {
  // (neg_a, A, neg_b, B, c_mod, C, reuse_a, reuse_b)
  return __builtin_amdgcn_wmma_f32_16x16x32_bf16(false, a, false, b, (short)0, c, false, false);
}

// Opaque zero offset: redefined by empty asm every call, so weight addresses
// become loop-variant (blocks LICM hoist+spill of B-fragments across the
// t-loop) WITHOUT touching the base pointers themselves — their global
// address-space provenance survives, keeping loads as global_load (not flat).
__device__ __forceinline__ size_t opq_zero() {
  size_t z = 0;
  asm volatile("" : "+s"(z));
  return z;
}

// A fragment (16x32 bf16) from LDS, row-major [16][stride].
// Lane L: row = L&15; K = {16*(L>>4) .. +15} within the 32-wide K-tile.
__device__ __forceinline__ v16bf ldsA(const __bf16* base, int stride, int kt, int lane) {
  const int m  = lane & 15;
  const int kb = kt * 32 + ((lane >> 4) << 4);
  const __bf16* p = base + m * stride + kb;
  v8bf lo = *(const v8bf*)p;
  v8bf hi = *(const v8bf*)(p + 8);
  return __builtin_shufflevector(lo, hi, 0,1,2,3,4,5,6,7,8,9,10,11,12,13,14,15);
}

// B fragment (32x16 bf16) from global row-major W[N][K] (B = W^T tile).
// Lane L: col n = n0 + (L&15); 16 contiguous K starting at kt*32 + 16*(L>>4).
__device__ __forceinline__ v16bf glbB(const __bf16* __restrict__ W, int K, int n0, int kt, int lane) {
  const int n = n0 + (lane & 15);
  const int k = kt * 32 + ((lane >> 4) << 4);
  return *(const v16bf*)(W + (size_t)n * K + k);
}

__device__ __forceinline__ float sigm(float x)  { return 1.0f / (1.0f + __expf(-x)); }
__device__ __forceinline__ float tanh_(float x) { float e = __expf(-2.0f * x); return (1.0f - e) / (1.0f + e); }

__global__ void f32_to_bf16(const float* __restrict__ s, __bf16* __restrict__ d, int n) {
  int i = blockIdx.x * blockDim.x + threadIdx.x;
  if (i < n) d[i] = (__bf16)s[i];
}

// Pipelined est accumulation: acc = h @ Wout^T tile for this wave.
__device__ __forceinline__ v8f est_acc(const __bf16* h_bf, const __bf16* __restrict__ Wout,
                                       int wv, int lane) {
  v8f acc = {};
  v16bf a_c = ldsA(h_bf, 520, 0, lane);
  v16bf b_c = glbB(Wout, DH, wv * 16, 0, lane);
  #pragma unroll 4
  for (int kt = 0; kt < 15; ++kt) {
    v16bf a_n = ldsA(h_bf, 520, kt + 1, lane);
    v16bf b_n = glbB(Wout, DH, wv * 16, kt + 1, lane);
    acc = wmma_bf(a_c, b_c, acc);
    a_c = a_n; b_c = b_n;
  }
  return wmma_bf(a_c, b_c, acc);
}

// One workgroup = 16 batch rows, 8 waves; wave wv owns hidden cols [64*wv, 64*wv+64).
// Loops over all T steps internally (batch rows are independent recurrences).
// Occupancy is structurally 2 waves/SIMD (1 WG per WGP), so allow ~512 VGPRs.
__global__ __launch_bounds__(256, 2) void brits_gru_kernel(
    const float* __restrict__ X, const float* __restrict__ Mk,
    const float* __restrict__ bih0, const float* __restrict__ bhh0,
    const float* __restrict__ bih1, const float* __restrict__ bhh1,
    const float* __restrict__ bout,
    const __bf16* __restrict__ Wih0, const __bf16* __restrict__ Whh0,
    const __bf16* __restrict__ Wih1, const __bf16* __restrict__ Whh1,
    const __bf16* __restrict__ Wout,
    float* __restrict__ out)
{
  extern __shared__ char smem[];
  __bf16* h_bf    = (__bf16*)smem;                 // [16][520] bf16 shadow of h
  __bf16* hmid_bf = h_bf + 16 * 520;               // [16][520] layer-0 output
  __bf16* imp_bf  = hmid_bf + 16 * 520;            // [16][136] imputed input

  const int tid  = threadIdx.x;
  const int lane = tid & 31;
  const int wv   = tid >> 5;           // wave 0..7
  const int mhi  = (lane >> 4) << 3;   // 0 or 8 (row offset of C/D half)
  const int nlo  = lane & 15;
  const int bb   = blockIdx.x * 16;    // batch tile base

  for (int i = tid; i < 16 * 520; i += 256) h_bf[i] = (__bf16)0.0f;

  // Per-lane, t-invariant bias constants (kept in VGPRs for the whole kernel).
  const float bO = bout[wv * 16 + nlo];
  float bR0[4], bZ0[4], bI0[4], bH0[4], bR1[4], bZ1[4], bI1[4], bH1[4];
  #pragma unroll
  for (int j = 0; j < 4; ++j) {
    const int n = wv * 64 + j * 16 + nlo;
    bR0[j] = bih0[n] + bhh0[n];
    bZ0[j] = bih0[512 + n] + bhh0[512 + n];
    bI0[j] = bih0[1024 + n];
    bH0[j] = bhh0[1024 + n];
    bR1[j] = bih1[n] + bhh1[n];
    bZ1[j] = bih1[512 + n] + bhh1[512 + n];
    bI1[j] = bih1[1024 + n];
    bH1[j] = bhh1[1024 + n];
  }

  float hreg[4][8];                    // fp32 master h at this wave's 64 cols
  #pragma unroll
  for (int j = 0; j < 4; ++j)
    #pragma unroll
    for (int e = 0; e < 8; ++e) hreg[j][e] = 0.0f;
  __syncthreads();

  const size_t OUT_OFF = (size_t)B_ * T_ * DIN;
  const size_t HF_OFF  = 2 * OUT_OFF;
  const size_t HS_OFF  = HF_OFF + (size_t)B_ * DH;
  float* pre = out;
  float* oc  = out + OUT_OFF;
  float* hf  = out + HF_OFF;
  float* hs  = out + HS_OFF;

  #pragma unroll 1
  for (int t = 0; t < T_; ++t) {
    // Opaque (actually-zero) offset, redefined each step: keeps every weight
    // load inside the loop (streamed from L2) instead of hoisted+spilled,
    // while preserving global address-space provenance of the bases.
    const size_t z = opq_zero();
    const __bf16* Wout_t = Wout + z;
    const __bf16* Wih0_t = Wih0 + z;
    const __bf16* Whh0_t = Whh0 + z;
    const __bf16* Wih1_t = Wih1 + z;
    const __bf16* Whh1_t = Whh1 + z;

    // ---- est = h @ Wout^T + bout ; imputed = m*x + (1-m)*est ----
    {
      v8f acc = est_acc(h_bf, Wout_t, wv, lane);
      const int n = wv * 16 + nlo;
      #pragma unroll
      for (int e = 0; e < 8; ++e) {
        const int m = e + mhi;
        const size_t idx = ((size_t)(bb + m) * T_ + t) * DIN + n;
        float est = acc[e] + bO;
        float xv = X[idx], mk = Mk[idx];
        float imp = mk * xv + (1.0f - mk) * est;
        pre[idx] = est;
        if (t > 0) oc[((size_t)(bb + m) * T_ + (t - 1)) * DIN + n] = imp;
        imp_bf[m * 136 + n] = (__bf16)imp;
      }
    }
    __syncthreads();

    float hm[4][8];
    // ---- GRU layer 0: gi from imputed (K=128), gh from h (K=512) ----
    #pragma unroll
    for (int j = 0; j < 4; ++j) {
      const int n0 = wv * 64 + j * 16;
      v8f ar = {}, az = {}, ai = {}, ah = {};
      {
        // gi: A = imputed, 4 K-tiles, software-pipelined
        v16bf a_c  = ldsA(imp_bf, 136, 0, lane);
        v16bf br_c = glbB(Wih0_t, DIN, n0,        0, lane);
        v16bf bz_c = glbB(Wih0_t, DIN, n0 + 512,  0, lane);
        v16bf bi_c = glbB(Wih0_t, DIN, n0 + 1024, 0, lane);
        #pragma unroll
        for (int kt = 0; kt < 3; ++kt) {
          v16bf a_n  = ldsA(imp_bf, 136, kt + 1, lane);
          v16bf br_n = glbB(Wih0_t, DIN, n0,        kt + 1, lane);
          v16bf bz_n = glbB(Wih0_t, DIN, n0 + 512,  kt + 1, lane);
          v16bf bi_n = glbB(Wih0_t, DIN, n0 + 1024, kt + 1, lane);
          ar = wmma_bf(a_c, br_c, ar);
          az = wmma_bf(a_c, bz_c, az);
          ai = wmma_bf(a_c, bi_c, ai);
          a_c = a_n; br_c = br_n; bz_c = bz_n; bi_c = bi_n;
        }
        ar = wmma_bf(a_c, br_c, ar);
        az = wmma_bf(a_c, bz_c, az);
        ai = wmma_bf(a_c, bi_c, ai);
      }
      {
        // gh: A = h, 16 K-tiles, software-pipelined
        v16bf a_c  = ldsA(h_bf, 520, 0, lane);
        v16bf br_c = glbB(Whh0_t, DH, n0,        0, lane);
        v16bf bz_c = glbB(Whh0_t, DH, n0 + 512,  0, lane);
        v16bf bh_c = glbB(Whh0_t, DH, n0 + 1024, 0, lane);
        #pragma unroll 5
        for (int kt = 0; kt < 15; ++kt) {
          v16bf a_n  = ldsA(h_bf, 520, kt + 1, lane);
          v16bf br_n = glbB(Whh0_t, DH, n0,        kt + 1, lane);
          v16bf bz_n = glbB(Whh0_t, DH, n0 + 512,  kt + 1, lane);
          v16bf bh_n = glbB(Whh0_t, DH, n0 + 1024, kt + 1, lane);
          ar = wmma_bf(a_c, br_c, ar);
          az = wmma_bf(a_c, bz_c, az);
          ah = wmma_bf(a_c, bh_c, ah);
          a_c = a_n; br_c = br_n; bz_c = bz_n; bh_c = bh_n;
        }
        ar = wmma_bf(a_c, br_c, ar);
        az = wmma_bf(a_c, bz_c, az);
        ah = wmma_bf(a_c, bh_c, ah);
      }
      #pragma unroll
      for (int e = 0; e < 8; ++e) {
        const int m = e + mhi;
        const int n = n0 + nlo;
        float r  = sigm(ar[e] + bR0[j]);
        float z2 = sigm(az[e] + bZ0[j]);
        float ng = tanh_(ai[e] + bI0[j] + r * (ah[e] + bH0[j]));
        float v  = (1.0f - z2) * ng + z2 * hreg[j][e];
        hm[j][e] = v;
        hmid_bf[m * 520 + n] = (__bf16)v;
      }
    }
    __syncthreads();

    // ---- GRU layer 1: input == hidden == hmid (K=512 both) ----
    #pragma unroll
    for (int j = 0; j < 4; ++j) {
      const int n0 = wv * 64 + j * 16;
      v8f ar = {}, az = {}, ai = {}, ah = {};
      {
        v16bf a_c  = ldsA(hmid_bf, 520, 0, lane);
        v16bf c0_c = glbB(Wih1_t, DH, n0,        0, lane);
        v16bf c1_c = glbB(Whh1_t, DH, n0,        0, lane);
        v16bf c2_c = glbB(Wih1_t, DH, n0 + 512,  0, lane);
        v16bf c3_c = glbB(Whh1_t, DH, n0 + 512,  0, lane);
        v16bf c4_c = glbB(Wih1_t, DH, n0 + 1024, 0, lane);
        v16bf c5_c = glbB(Whh1_t, DH, n0 + 1024, 0, lane);
        #pragma unroll 5
        for (int kt = 0; kt < 15; ++kt) {
          v16bf a_n  = ldsA(hmid_bf, 520, kt + 1, lane);
          v16bf c0_n = glbB(Wih1_t, DH, n0,        kt + 1, lane);
          v16bf c1_n = glbB(Whh1_t, DH, n0,        kt + 1, lane);
          v16bf c2_n = glbB(Wih1_t, DH, n0 + 512,  kt + 1, lane);
          v16bf c3_n = glbB(Whh1_t, DH, n0 + 512,  kt + 1, lane);
          v16bf c4_n = glbB(Wih1_t, DH, n0 + 1024, kt + 1, lane);
          v16bf c5_n = glbB(Whh1_t, DH, n0 + 1024, kt + 1, lane);
          ar = wmma_bf(a_c, c0_c, ar);
          ar = wmma_bf(a_c, c1_c, ar);
          az = wmma_bf(a_c, c2_c, az);
          az = wmma_bf(a_c, c3_c, az);
          ai = wmma_bf(a_c, c4_c, ai);
          ah = wmma_bf(a_c, c5_c, ah);
          a_c = a_n; c0_c = c0_n; c1_c = c1_n; c2_c = c2_n;
          c3_c = c3_n; c4_c = c4_n; c5_c = c5_n;
        }
        ar = wmma_bf(a_c, c0_c, ar);
        ar = wmma_bf(a_c, c1_c, ar);
        az = wmma_bf(a_c, c2_c, az);
        az = wmma_bf(a_c, c3_c, az);
        ai = wmma_bf(a_c, c4_c, ai);
        ah = wmma_bf(a_c, c5_c, ah);
      }
      #pragma unroll
      for (int e = 0; e < 8; ++e) {
        const int m = e + mhi;
        const int n = n0 + nlo;
        float r  = sigm(ar[e] + bR1[j]);
        float z2 = sigm(az[e] + bZ1[j]);
        float ng = tanh_(ai[e] + bI1[j] + r * (ah[e] + bH1[j]));
        float v  = (1.0f - z2) * ng + z2 * hm[j][e];
        hreg[j][e] = v;
        h_bf[m * 520 + n] = (__bf16)v;
        hs[((size_t)(bb + m) * T_ + t) * DH + n] = v;
      }
    }
    __syncthreads();
  }

  // final est -> output_collector[:, T-1, :]
  {
    v8f acc = est_acc(h_bf, Wout, wv, lane);
    const int n = wv * 16 + nlo;
    #pragma unroll
    for (int e = 0; e < 8; ++e) {
      const int m = e + mhi;
      oc[((size_t)(bb + m) * T_ + (T_ - 1)) * DIN + n] = acc[e] + bO;
    }
  }
  // h_final
  #pragma unroll
  for (int j = 0; j < 4; ++j) {
    const int n = wv * 64 + j * 16 + nlo;
    #pragma unroll
    for (int e = 0; e < 8; ++e) {
      const int m = e + mhi;
      hf[(size_t)(bb + m) * DH + n] = hreg[j][e];
    }
  }
}

extern "C" void kernel_launch(void* const* d_in, const int* in_sizes, int n_in,
                              void* d_out, int out_size, void* d_ws, size_t ws_size,
                              hipStream_t stream) {
  (void)in_sizes; (void)n_in; (void)out_size; (void)ws_size;
  const float* X    = (const float*)d_in[0];
  const float* Mk   = (const float*)d_in[1];
  const float* Wih0 = (const float*)d_in[2];
  const float* Whh0 = (const float*)d_in[3];
  const float* bih0 = (const float*)d_in[4];
  const float* bhh0 = (const float*)d_in[5];
  const float* Wih1 = (const float*)d_in[6];
  const float* Whh1 = (const float*)d_in[7];
  const float* bih1 = (const float*)d_in[8];
  const float* bhh1 = (const float*)d_in[9];
  const float* Wout = (const float*)d_in[10];
  const float* bout = (const float*)d_in[11];

  // Workspace layout: bf16 copies of all weight matrices (row-major [N][K]).
  __bf16* Wih0b = (__bf16*)d_ws;                 // 1536*128
  __bf16* Whh0b = Wih0b + 1536 * 128;            // 1536*512
  __bf16* Wih1b = Whh0b + 1536 * 512;            // 1536*512
  __bf16* Whh1b = Wih1b + 1536 * 512;            // 1536*512
  __bf16* Woutb = Whh1b + 1536 * 512;            // 128*512

  f32_to_bf16<<<(1536 * 128 + 255) / 256, 256, 0, stream>>>(Wih0, Wih0b, 1536 * 128);
  f32_to_bf16<<<(1536 * 512 + 255) / 256, 256, 0, stream>>>(Whh0, Whh0b, 1536 * 512);
  f32_to_bf16<<<(1536 * 512 + 255) / 256, 256, 0, stream>>>(Wih1, Wih1b, 1536 * 512);
  f32_to_bf16<<<(1536 * 512 + 255) / 256, 256, 0, stream>>>(Whh1, Whh1b, 1536 * 512);
  f32_to_bf16<<<(128 * 512 + 255) / 256, 256, 0, stream>>>(Wout, Woutb, 128 * 512);

  const size_t smem = (size_t)(16 * 520 * 2) * 2   // h_bf + hmid_bf
                    + (size_t)(16 * 136 * 2);      // imp_bf
  brits_gru_kernel<<<dim3(B_ / 16), dim3(256), smem, stream>>>(
      X, Mk, bih0, bhh0, bih1, bhh1, bout,
      Wih0b, Whh0b, Wih1b, Whh1b, Woutb, (float*)d_out);
}
